// VNGNN_59004260712936
// MI455X (gfx1250) — matrix operand order repaired and checked by reading, compiled, not run on hardware
//
#include <hip/hip_runtime.h>
#include <hip/hip_bf16.h>
#include <math.h>

#define DIM   256
#define NN    200000
#define CC    1000
#define WMAT  (DIM * DIM)          // 65536 elements per 256x256 weight
#define NSTRIPS ((NN + 15) / 16)   // 12500, exact

typedef __bf16 bf16_t;
typedef __attribute__((ext_vector_type(16))) __bf16 v16bf;
typedef __attribute__((ext_vector_type(8)))  float  v8f;

// ---------------------------------------------------------------------------
// helpers: straight-line transcendentals (no libm branchy paths)
// ---------------------------------------------------------------------------
__device__ inline float fast_tanh(float x) {
#if __has_builtin(__builtin_amdgcn_tanhf)
    return __builtin_amdgcn_tanhf(x);        // native v_tanh_f32 on gfx1250
#else
    const float ex = __expf(2.0f * x);       // v_exp_f32
    return 1.0f - __fdividef(2.0f, ex + 1.0f);
#endif
}

__device__ inline float fast_sigmoid(float x) {
    return __fdividef(1.0f, 1.0f + __expf(-x));
}

__device__ inline void atomicMaxF(float* addr, float val) {
    int old = __float_as_int(*addr);
    while (__int_as_float(old) < val) {
        int assumed = old;
        old = atomicCAS((int*)addr, assumed, __float_as_int(val));
        if (old == assumed) break;
    }
}

// ---------------------------------------------------------------------------
// K0a: convert the two big weights to bf16 AND pre-swizzle into WMMA
// B-fragment-major layout, so LDS fragment loads are 2x ds_load_b128.
//
// Fragment index: hn(2) x nt(8) x kc(8); each fragment = 32 lanes x 16 elems.
//   flat i: e=i&15, lane=(i>>4)&31, kc=(i>>9)&7, nt=(i>>12)&7, hn=(i>>15)&1
//   source: K = kc*32 + ((lane>=16)?8:0) + (e&7) + ((e>>3)*16) ; col = hn*128 + nt*16 + (lane&15)
// ---------------------------------------------------------------------------
__global__ void prep_weights(const float* __restrict__ Wk, const float* __restrict__ H1w,
                             bf16_t* __restrict__ wkb, bf16_t* __restrict__ h1wb) {
    const int i = blockIdx.x * 256 + threadIdx.x;
    if (i >= WMAT) return;
    const int e    = i & 15;
    const int lane = (i >> 4) & 31;
    const int kc   = (i >> 9) & 7;
    const int nt   = (i >> 12) & 7;
    const int hn   = (i >> 15) & 1;
    const int k    = kc * 32 + ((lane >> 4) << 3) + (e & 7) + ((e >> 3) << 4);
    const int col  = hn * 128 + nt * 16 + (lane & 15);
    const int src  = k * DIM + col;
    wkb[i]  = (bf16_t)Wk[src];
    h1wb[i] = (bf16_t)H1w[src];
}

// K0b: init segment-reduction buffers (workspace is poisoned by harness)
__global__ void init_red(float* __restrict__ m, float* __restrict__ denom,
                         float* __restrict__ ctx) {
    int i = blockIdx.x * 256 + threadIdx.x;
    if (i < CC) { m[i] = -3.0e38f; denom[i] = 0.0f; }
    if (i < CC * DIM) ctx[i] = 0.0f;
}

// ---------------------------------------------------------------------------
// K1a: q = g @ Wq ; h_trans = tanh(g @ Ws + bs)      (C x D, tiny)
// ---------------------------------------------------------------------------
__global__ void cluster_qh(const float* __restrict__ g, const float* __restrict__ Wq,
                           const float* __restrict__ Ws, const float* __restrict__ bs,
                           float* __restrict__ q, float* __restrict__ h_trans) {
    const int c = blockIdx.x, d = threadIdx.x;
    __shared__ float gr[DIM];
    gr[d] = g[c * DIM + d];
    __syncthreads();
    float aq = 0.f, as_ = 0.f;
#pragma unroll 8
    for (int k = 0; k < DIM; ++k) {
        const float gv = gr[k];
        aq  = fmaf(gv, Wq[k * DIM + d], aq);
        as_ = fmaf(gv, Ws[k * DIM + d], as_);
    }
    q[c * DIM + d]       = aq;
    h_trans[c * DIM + d] = fast_tanh(as_ + bs[d]);
}

// K1b: u = h_trans @ G1w + G1b + H1b                 (C x D, tiny)
__global__ void cluster_u(const float* __restrict__ h_trans, const float* __restrict__ G1w,
                          const float* __restrict__ G1b, const float* __restrict__ H1b,
                          float* __restrict__ u) {
    const int c = blockIdx.x, d = threadIdx.x;
    __shared__ float hr[DIM];
    hr[d] = h_trans[c * DIM + d];
    __syncthreads();
    float acc = 0.f;
#pragma unroll 8
    for (int k = 0; k < DIM; ++k) acc = fmaf(hr[k], G1w[k * DIM + d], acc);
    u[c * DIM + d] = acc + G1b[d] + H1b[d];
}

// ---------------------------------------------------------------------------
// K2: fused node pass. One wave per 16-row strip of h.
//   GEMM1: k = h @ Wk  -> s = tanh(k + q[seg] + b_attn) . w_score + b_score
//          -> store s, atomicMax into per-cluster m
//   GEMM2: hw = h @ H1w -> z1 = sigmoid(hw + u[seg])
//          -> out = (1-z1)*h + z1*h_trans[seg]
// A-fragments loaded once (float4 + bf16 cvt), reused for both GEMMs.
// B weights staged through 64KB LDS per column-half, already fragment-major.
// ---------------------------------------------------------------------------
__device__ inline void stage_half(const bf16_t* __restrict__ src, bf16_t* dst,
                                  int hn, int tid) {
    // copy 32768 bf16 (64KB), contiguous both sides, 16B vectors
    const uint4* s = (const uint4*)(src + hn * 32768);
    uint4* d = (uint4*)dst;
    for (int i = tid; i < 4096; i += 128) d[i] = s[i];
}

__global__ __launch_bounds__(128)
void node_fused(const float* __restrict__ h, const int* __restrict__ seg,
                const bf16_t* __restrict__ wkb, const bf16_t* __restrict__ h1wb,
                const float* __restrict__ q, const float* __restrict__ b_attn,
                const float* __restrict__ w_score, const float* __restrict__ b_score,
                const float* __restrict__ u, const float* __restrict__ h_trans,
                float* __restrict__ s_out, float* __restrict__ m_out,
                float* __restrict__ out) {
    __shared__ __align__(32) bf16_t lds_w[32768];   // 64KB: one column-half, fragment-major

    const int tid   = threadIdx.x;
    const int lane  = tid & 31;
    const int wave  = tid >> 5;
    const int strip = blockIdx.x * 4 + wave;
    const bool active = (strip < NSTRIPS);
    const int r0    = strip * 16;
    const int half  = lane >> 4;          // 0: lanes 0-15, 1: lanes 16-31
    const int l15   = lane & 15;

    // prefetch GEMM2 weight fragments while we work on GEMM1
    __builtin_prefetch(h1wb + tid * 256, 0, 3);
    __builtin_prefetch(h1wb + 32768 + tid * 256, 0, 3);

    // per-wave loop invariants: cluster offset of each of this lane's 8 rows
    int cseg[8];
#pragma unroll
    for (int r = 0; r < 8; ++r)
        cseg[r] = seg[min(r0 + half * 8 + r, NN - 1)] * DIM;

    // ---- A fragments: 16 rows x 256 cols of h as bf16, ISA 16-bit A layout ----
    // lanes 0-15: row=lane, K = kc*32 + {0..7, 16..23}
    // lanes 16-31: row=lane-16, K = kc*32 + {8..15, 24..31}
    v16bf a[8];
    {
        const int arow = min(r0 + l15, NN - 1);
        const float4* hrow4 = (const float4*)(h + (size_t)arow * DIM);
#pragma unroll
        for (int kc = 0; kc < 8; ++kc) {
            const int b0 = kc * 8 + half * 2;
            const float4 p0 = hrow4[b0];          // K offsets +0..3
            const float4 p1 = hrow4[b0 + 1];      // +4..7
            const float4 p2 = hrow4[b0 + 4];      // +16..19
            const float4 p3 = hrow4[b0 + 5];      // +20..23
            a[kc][0]  = (bf16_t)p0.x; a[kc][1]  = (bf16_t)p0.y;
            a[kc][2]  = (bf16_t)p0.z; a[kc][3]  = (bf16_t)p0.w;
            a[kc][4]  = (bf16_t)p1.x; a[kc][5]  = (bf16_t)p1.y;
            a[kc][6]  = (bf16_t)p1.z; a[kc][7]  = (bf16_t)p1.w;
            a[kc][8]  = (bf16_t)p2.x; a[kc][9]  = (bf16_t)p2.y;
            a[kc][10] = (bf16_t)p2.z; a[kc][11] = (bf16_t)p2.w;
            a[kc][12] = (bf16_t)p3.x; a[kc][13] = (bf16_t)p3.y;
            a[kc][14] = (bf16_t)p3.z; a[kc][15] = (bf16_t)p3.w;
        }
    }

    // =================== GEMM1: attention scores ===================
    // per-lane partial score dot; butterfly-reduced ONCE after all tiles
    float sacc[8];
#pragma unroll
    for (int r = 0; r < 8; ++r) sacc[r] = 0.f;

    for (int hn = 0; hn < 2; ++hn) {
        __syncthreads();
        stage_half(wkb, lds_w, hn, tid);
        __syncthreads();
        if (active) {
            for (int nt = 0; nt < 8; ++nt) {
                const int col = hn * 128 + nt * 16 + l15;    // global output column
                v8f acc = {0.f, 0.f, 0.f, 0.f, 0.f, 0.f, 0.f, 0.f};
#pragma unroll
                for (int kc = 0; kc < 8; ++kc) {
                    // fragment-major LDS: lane's 16 elems contiguous -> 2x ds_load_b128
                    const v16bf b = *(const v16bf*)(lds_w + (nt * 8 + kc) * 512 + lane * 16);
                    acc = __builtin_amdgcn_wmma_f32_16x16x32_bf16(
                              false, a[kc], false, b, (short)0, acc, false, false);
                }
                const float ba  = b_attn[col];
                const float wsc = w_score[col];
#pragma unroll
                for (int r = 0; r < 8; ++r) {
                    const float t = fast_tanh(acc[r] + q[cseg[r] + col] + ba);
                    sacc[r] = fmaf(t, wsc, sacc[r]);
                }
            }
        }
    }
    // one butterfly per row: 16-lane half sum -> every lane of the half holds total
#pragma unroll
    for (int r = 0; r < 8; ++r) {
        float t = sacc[r];
        t += __shfl_xor(t, 1, 32);
        t += __shfl_xor(t, 2, 32);
        t += __shfl_xor(t, 4, 32);
        t += __shfl_xor(t, 8, 32);
        sacc[r] = t;
    }
    if (active && l15 == 0) {
        const float bsc = *b_score;
#pragma unroll
        for (int r = 0; r < 8; ++r) {
            const int row = r0 + half * 8 + r;
            const float sv = sacc[r] + bsc;
            s_out[row] = sv;
            atomicMaxF(&m_out[cseg[r] >> 8], sv);   // cseg = cluster*DIM, DIM=256
        }
    }

    // =================== GEMM2: warp-gate merge ===================
    for (int hn = 0; hn < 2; ++hn) {
        __syncthreads();
        stage_half(h1wb, lds_w, hn, tid);
        __syncthreads();
        if (active) {
            for (int nt = 0; nt < 8; ++nt) {
                const int col = hn * 128 + nt * 16 + l15;
                v8f acc = {0.f, 0.f, 0.f, 0.f, 0.f, 0.f, 0.f, 0.f};
#pragma unroll
                for (int kc = 0; kc < 8; ++kc) {
                    const v16bf b = *(const v16bf*)(lds_w + (nt * 8 + kc) * 512 + lane * 16);
                    acc = __builtin_amdgcn_wmma_f32_16x16x32_bf16(
                              false, a[kc], false, b, (short)0, acc, false, false);
                }
#pragma unroll
                for (int r = 0; r < 8; ++r) {
                    const int row = r0 + half * 8 + r;
                    const float z  = fast_sigmoid(acc[r] + u[cseg[r] + col]);
                    const float hv = h[(size_t)row * DIM + col];
                    const float ht = h_trans[cseg[r] + col];
                    out[(size_t)row * DIM + col] = (1.f - z) * hv + z * ht;
                }
            }
        }
    }
}

// ---------------------------------------------------------------------------
// K3: segmented softmax accumulation: e = exp(s - m[seg]); denom += e; ctx += e*h
// one 32-lane group per node
// ---------------------------------------------------------------------------
__global__ void ctx_accum(const float* __restrict__ h, const float* __restrict__ s,
                          const float* __restrict__ m, const int* __restrict__ seg,
                          float* __restrict__ denom, float* __restrict__ ctx) {
    const int node = blockIdx.x * 8 + (threadIdx.x >> 5);
    const int lane = threadIdx.x & 31;
    if (node >= NN) return;
    const int c = seg[node];
    const float e = __expf(s[node] - m[c]);
    if (lane == 0) atomicAdd(&denom[c], e);
    const float* hrow = h + (size_t)node * DIM;
#pragma unroll
    for (int j = 0; j < 8; ++j) {
        const int d = lane + 32 * j;
        atomicAdd(&ctx[c * DIM + d], e * hrow[d]);
    }
}

// ---------------------------------------------------------------------------
// K4a: g_trans = tanh((ctx/denom) @ Wv + bv)          (C x D, tiny)
// ---------------------------------------------------------------------------
__global__ void vn_gtrans(const float* __restrict__ ctx, const float* __restrict__ denom,
                          const float* __restrict__ Wv, const float* __restrict__ bv,
                          float* __restrict__ g_trans) {
    const int c = blockIdx.x, d = threadIdx.x;
    __shared__ float cr[DIM];
    cr[d] = ctx[c * DIM + d];
    __syncthreads();
    float acc = 0.f;
#pragma unroll 8
    for (int k = 0; k < DIM; ++k) acc = fmaf(cr[k], Wv[k * DIM + d], acc);
    g_trans[c * DIM + d] = fast_tanh(__fdividef(acc, denom[c]) + bv[d]);
}

// K4b: z2 = sigmoid(g_trans@H2w + H2b + g_hat@G2w + G2b);
//      out[N+c] = (1-z2)*g_trans + z2*g_hat
__global__ void vn_merge(const float* __restrict__ g_trans, const float* __restrict__ g_hat,
                         const float* __restrict__ H2w, const float* __restrict__ H2b,
                         const float* __restrict__ G2w, const float* __restrict__ G2b,
                         float* __restrict__ out) {
    const int c = blockIdx.x, d = threadIdx.x;
    __shared__ float gt[DIM];
    __shared__ float gh[DIM];
    gt[d] = g_trans[c * DIM + d];
    gh[d] = g_hat[c * DIM + d];
    __syncthreads();
    float a1 = 0.f, a2 = 0.f;
#pragma unroll 8
    for (int k = 0; k < DIM; ++k) {
        a1 = fmaf(gt[k], H2w[k * DIM + d], a1);
        a2 = fmaf(gh[k], G2w[k * DIM + d], a2);
    }
    const float z = fast_sigmoid(a1 + H2b[d] + a2 + G2b[d]);
    out[(size_t)NN * DIM + c * DIM + d] = (1.f - z) * gt[d] + z * gh[d];
}

// ---------------------------------------------------------------------------
// entry
// ---------------------------------------------------------------------------
extern "C" void kernel_launch(void* const* d_in, const int* in_sizes, int n_in,
                              void* d_out, int out_size, void* d_ws, size_t ws_size,
                              hipStream_t stream) {
    const float* h       = (const float*)d_in[0];
    const float* g       = (const float*)d_in[1];
    const float* g_hat   = (const float*)d_in[2];
    const int*   seg     = (const int*)  d_in[3];
    const float* Wq      = (const float*)d_in[4];
    const float* Wk      = (const float*)d_in[5];
    const float* b_attn  = (const float*)d_in[6];
    const float* w_score = (const float*)d_in[7];
    const float* b_score = (const float*)d_in[8];
    const float* Wv      = (const float*)d_in[9];
    const float* bv      = (const float*)d_in[10];
    const float* Ws      = (const float*)d_in[11];
    const float* bs      = (const float*)d_in[12];
    const float* H1w     = (const float*)d_in[13];
    const float* H1b     = (const float*)d_in[14];
    const float* G1w     = (const float*)d_in[15];
    const float* G1b     = (const float*)d_in[16];
    const float* H2w     = (const float*)d_in[17];
    const float* H2b     = (const float*)d_in[18];
    const float* G2w     = (const float*)d_in[19];
    const float* G2b     = (const float*)d_in[20];
    float* out = (float*)d_out;

    // workspace carve (256B aligned)
    char* ws = (char*)d_ws;
    size_t off = 0;
    auto carve = [&](size_t bytes) -> void* {
        void* p = ws + off;
        off = (off + bytes + 255) & ~(size_t)255;
        return p;
    };
    bf16_t* wkb     = (bf16_t*)carve(WMAT * sizeof(bf16_t));   // 128KB, fragment-major
    bf16_t* h1wb    = (bf16_t*)carve(WMAT * sizeof(bf16_t));   // 128KB, fragment-major
    float*  qbuf    = (float*) carve(CC * DIM * sizeof(float)); // 1MB
    float*  h_trans = (float*) carve(CC * DIM * sizeof(float));
    float*  ubuf    = (float*) carve(CC * DIM * sizeof(float));
    float*  gtr     = (float*) carve(CC * DIM * sizeof(float));
    float*  ctx     = (float*) carve(CC * DIM * sizeof(float));
    float*  sbuf    = (float*) carve(NN * sizeof(float));       // 800KB
    float*  mbuf    = (float*) carve(CC * sizeof(float));
    float*  dbuf    = (float*) carve(CC * sizeof(float));
    (void)ws_size; (void)n_in; (void)in_sizes; (void)out_size;

    // 0) one-time converts (+ fragment swizzle) + reduction init
    prep_weights<<<(WMAT + 255) / 256, 256, 0, stream>>>(Wk, H1w, wkb, h1wb);
    init_red<<<(CC * DIM + 255) / 256, 256, 0, stream>>>(mbuf, dbuf, ctx);

    // 1) cluster-side prep (tiny GEMMs)
    cluster_qh<<<CC, DIM, 0, stream>>>(g, Wq, Ws, bs, qbuf, h_trans);
    cluster_u<<<CC, DIM, 0, stream>>>(h_trans, G1w, G1b, H1b, ubuf);

    // 2) fused WMMA node pass (both big GEMMs, one read of h)
    node_fused<<<(NSTRIPS + 3) / 4, 128, 0, stream>>>(
        h, seg, wkb, h1wb, qbuf, b_attn, w_score, b_score,
        ubuf, h_trans, sbuf, mbuf, out);

    // 3) segmented softmax accumulation
    ctx_accum<<<(NN + 7) / 8, 256, 0, stream>>>(h, sbuf, mbuf, seg, dbuf, ctx);

    // 4) virtual-node finalize
    vn_gtrans<<<CC, DIM, 0, stream>>>(ctx, dbuf, Wv, bv, gtr);
    vn_merge<<<CC, DIM, 0, stream>>>(gtr, g_hat, H2w, H2b, G2w, G2b, out);
}
